// GraphSAGE_15625091022994
// MI455X (gfx1250) — matrix operand order repaired and checked by reading, not compiled, not run on hardware
//
#include <hip/hip_runtime.h>
#include <hip/hip_bf16.h>

typedef __attribute__((ext_vector_type(16))) __bf16 v16bf;
typedef __attribute__((ext_vector_type(4)))  __bf16 v4bf;
typedef __attribute__((ext_vector_type(8)))  float  v8f;

#define TPB 256

// ---------- fragment loaders (bf16-resident operands; pure b128 loads) ----------
// A fragment 16x32 bf16 (ISA 7.12.2): lane row m = mBase+(lane&15), g = lane>>4.
// elements 0..15 hold K = kb+16g+{0..15} (contiguous, matching B fragment).
static __device__ __forceinline__ v16bf load_a_frag(const __bf16* __restrict__ row,
                                                    int kb, int g) {
    union { v16bf v; uint4 q[2]; } u;
    u.q[0] = *(const uint4*)(row + kb + 16 * g);
    u.q[1] = *(const uint4*)(row + kb + 16 * g + 8);
    return u.v;
}

// B fragment 32x16 bf16: Wt is N-major (Wt[n*K1 + k]). lane column n = nBase+(lane&15);
// lanes 0-15 hold K = kb+{0..15}, lanes 16-31 hold K = kb+16+{0..15} (contiguous 32B).
static __device__ __forceinline__ v16bf load_b_frag(const __bf16* __restrict__ Wt,
                                                    int K1, int nBase, int kb, int lane) {
    const __bf16* colp = Wt + (size_t)(nBase + (lane & 15)) * K1 + kb + 16 * (lane >> 4);
    union { v16bf v; uint4 q[2]; } u;
    u.q[0] = *(const uint4*)(colp);
    u.q[1] = *(const uint4*)(colp + 8);
    return u.v;
}

// ---------- small kernels ----------
__global__ void deg_kernel(const long long* __restrict__ dst, float* __restrict__ cnt, int E) {
    int e = blockIdx.x * blockDim.x + threadIdx.x;
    if (e < E) atomicAdd(&cnt[dst[e]], 1.0f);
}

__global__ void inv_kernel(float* __restrict__ c, int n) {
    int i = blockIdx.x * blockDim.x + threadIdx.x;
    if (i < n) c[i] = 1.0f / fmaxf(c[i], 1.0f);
}

// f32 (K,N) row-major -> bf16 N-major transpose (weights; tiny)
__global__ void wt_convert(const float* __restrict__ W, __bf16* __restrict__ Wt, int K, int N) {
    int t = blockIdx.x * blockDim.x + threadIdx.x;
    if (t >= K * N) return;
    int k = t / N, n = t - k * N;
    Wt[(size_t)n * K + k] = (__bf16)W[t];
}

// plain f32 -> bf16 copy, float4-granular
__global__ void cvt_bf16_kernel(const float* __restrict__ in, __bf16* __restrict__ out,
                                long long n4) {
    long long t = (long long)blockIdx.x * blockDim.x + threadIdx.x;
    if (t >= n4) return;
    const float4 v = ((const float4*)in)[t];
    v4bf o = { (__bf16)v.x, (__bf16)v.y, (__bf16)v.z, (__bf16)v.w };
    *(v4bf*)(out + 4 * t) = o;
}

// mean = agg * inv[node], stored bf16
template <int D>
__global__ void meanbf_kernel(const float* __restrict__ agg, const float* __restrict__ inv,
                              __bf16* __restrict__ mbf, int nodes) {
    constexpr int Q = D / 4;
    int t = blockIdx.x * blockDim.x + threadIdx.x;
    if (t >= nodes * Q) return;
    int node = t / Q, off = (t - node * Q) * 4;
    float s = inv[node];
    const float4 v = *(const float4*)(agg + (size_t)node * D + off);
    v4bf o = { (__bf16)(v.x * s), (__bf16)(v.y * s), (__bf16)(v.z * s), (__bf16)(v.w * s) };
    *(v4bf*)(mbf + (size_t)node * D + off) = o;
}

// one wave per edge; lane l handles features 4l..4l+3 (coalesced 512B row gather)
__global__ void scatter128_kernel(const float* __restrict__ feat, const long long* __restrict__ src,
                                  const long long* __restrict__ dst, float* __restrict__ agg, int E) {
    int tid = blockIdx.x * blockDim.x + threadIdx.x;
    int e = tid >> 5, lane = tid & 31;
    if (e >= E) return;
    long long s = src[e], d = dst[e];
    const float4 v = ((const float4*)(feat + (size_t)s * 128))[lane];
    float* a = agg + (size_t)d * 128 + lane * 4;
    atomicAdd(a + 0, v.x); atomicAdd(a + 1, v.y);
    atomicAdd(a + 2, v.z); atomicAdd(a + 3, v.w);
}

__global__ void scatter64_kernel(const float* __restrict__ feat, const long long* __restrict__ src,
                                 const long long* __restrict__ dst, float* __restrict__ agg, int E) {
    int tid = blockIdx.x * blockDim.x + threadIdx.x;
    int e = tid >> 4, lane = tid & 15;
    if (e >= E) return;
    long long s = src[e], d = dst[e];
    const float4 v = ((const float4*)(feat + (size_t)s * 64))[lane];
    float* a = agg + (size_t)d * 64 + lane * 4;
    atomicAdd(a + 0, v.x); atomicAdd(a + 1, v.y);
    atomicAdd(a + 2, v.z); atomicAdd(a + 3, v.w);
}

// H = tanh( mean @ Wl + X @ Wr + b ). One 16x32 output tile per wave: two
// independent accumulators alternate so consecutive v_wmma ops have no RAW chain.
// Stores f32 (for next scatter) and bf16 (next layer's A-operand).
template <int K1, int N>
__global__ __launch_bounds__(TPB)
void sage_gemm_kernel(const __bf16* __restrict__ meanbf, const __bf16* __restrict__ xbf,
                      const __bf16* __restrict__ WlT, const __bf16* __restrict__ WrT,
                      const float* __restrict__ bias, float* __restrict__ outf,
                      __bf16* __restrict__ outbf, int nodes) {
    constexpr int NP = N / 32;                       // n-tile pairs
    const int wid  = (blockIdx.x * blockDim.x + threadIdx.x) >> 5;
    const int lane = threadIdx.x & 31;
    const int mTile = wid / NP;
    const int nBase = (wid % NP) * 32;
    const int mBase = mTile * 16;
    if (mBase >= nodes) return;                      // wave-uniform; EXEC stays all-ones

    const int g    = lane >> 4;
    const int mrow = mBase + (lane & 15);
    const __bf16* mRow = meanbf + (size_t)mrow * K1;
    const __bf16* xRow = xbf   + (size_t)mrow * K1;

    v8f acc0 = {}, acc1 = {};
#pragma unroll
    for (int kb = 0; kb < K1; kb += 32) {
        v16bf a1  = load_a_frag(mRow, kb, g);
        v16bf b1a = load_b_frag(WlT, K1, nBase,      kb, lane);
        v16bf b1b = load_b_frag(WlT, K1, nBase + 16, kb, lane);
        acc0 = __builtin_amdgcn_wmma_f32_16x16x32_bf16(false, a1, false, b1a,
                                                       (short)0, acc0, false, false);
        acc1 = __builtin_amdgcn_wmma_f32_16x16x32_bf16(false, a1, false, b1b,
                                                       (short)0, acc1, false, false);
        v16bf a2  = load_a_frag(xRow, kb, g);
        v16bf b2a = load_b_frag(WrT, K1, nBase,      kb, lane);
        v16bf b2b = load_b_frag(WrT, K1, nBase + 16, kb, lane);
        acc0 = __builtin_amdgcn_wmma_f32_16x16x32_bf16(false, a2, false, b2a,
                                                       (short)0, acc0, false, false);
        acc1 = __builtin_amdgcn_wmma_f32_16x16x32_bf16(false, a2, false, b2b,
                                                       (short)0, acc1, false, false);
    }

    // C/D layout: VGPR r -> row mBase + 8*g + r, column (lane&15) of each 16-wide tile
    const int ncol = lane & 15;
    const float bv0 = bias[nBase + ncol];
    const float bv1 = bias[nBase + 16 + ncol];
    const size_t rowOff = (size_t)(mBase + 8 * g) * N;
    float*  of = outf  + rowOff;
    __bf16* ob = outbf + rowOff;
#pragma unroll
    for (int r = 0; r < 8; ++r) {
        const size_t ro = (size_t)r * N;
        float t0 = tanhf(acc0[r] + bv0);
        float t1 = tanhf(acc1[r] + bv1);
        of[ro + nBase + ncol]      = t0;
        of[ro + nBase + 16 + ncol] = t1;
        ob[ro + nBase + ncol]      = (__bf16)t0;
        ob[ro + nBase + 16 + ncol] = (__bf16)t1;
    }
}

// out[node, j] = h3[node,:] . Wlin[:, j] + blin[j]   (64 -> 10, VALU)
__global__ void head_kernel(const float* __restrict__ h, const float* __restrict__ Wlin,
                            const float* __restrict__ blin, float* __restrict__ out, int nodes) {
    int t = blockIdx.x * blockDim.x + threadIdx.x;
    if (t >= nodes * 10) return;
    int node = t / 10, j = t - node * 10;
    const float* hr = h + (size_t)node * 64;
    float acc = blin[j];
#pragma unroll
    for (int k = 0; k < 64; ++k) acc += hr[k] * Wlin[k * 10 + j];
    out[t] = acc;
}

// ---------- launch ----------
static inline size_t align256(size_t v) { return (v + 255) & ~(size_t)255; }
static inline int cdiv(long long a, int b) { return (int)((a + b - 1) / b); }

extern "C" void kernel_launch(void* const* d_in, const int* in_sizes, int n_in,
                              void* d_out, int out_size, void* d_ws, size_t ws_size,
                              hipStream_t stream) {
    const float*     x    = (const float*)d_in[0];
    const long long* eidx = (const long long*)d_in[1];
    const float *W1l = (const float*)d_in[2], *W1r = (const float*)d_in[3], *b1 = (const float*)d_in[4];
    const float *W2l = (const float*)d_in[5], *W2r = (const float*)d_in[6], *b2 = (const float*)d_in[7];
    const float *W3l = (const float*)d_in[8], *W3r = (const float*)d_in[9], *b3 = (const float*)d_in[10];
    const float *Wlin = (const float*)d_in[11], *blin = (const float*)d_in[12];
    float* out = (float*)d_out;

    const int nodes = in_sizes[0] / 128;          // 100000
    const int E     = in_sizes[1] / 2;            // 1600000
    const long long* src = eidx;
    const long long* dst = eidx + E;

    // workspace carve-up
    char* ws = (char*)d_ws;
    size_t o = 0;
    float* inv = (float*)(ws + o);                 o += align256((size_t)nodes * 4);
    __bf16* wt = (__bf16*)(ws + o);                o += align256(6 * 8192 * sizeof(__bf16));
    __bf16 *wt1l = wt, *wt1r = wt + 8192, *wt2l = wt + 2 * 8192,
           *wt2r = wt + 3 * 8192, *wt3l = wt + 4 * 8192, *wt3r = wt + 5 * 8192;
    float*  agg  = (float*)(ws + o);               o += align256((size_t)nodes * 128 * 4);
    __bf16* mbf  = (__bf16*)(ws + o);              o += align256((size_t)nodes * 128 * 2);
    __bf16* xbf  = (__bf16*)(ws + o);              o += align256((size_t)nodes * 128 * 2); // then h2 bf16
    __bf16* hAbf = (__bf16*)(ws + o);              o += align256((size_t)nodes * 64 * 2);  // h1/h3 bf16
    float*  hA   = (float*)(ws + o);               o += align256((size_t)nodes * 64 * 4);  // h1, then h3
    float*  hB   = (float*)(ws + o);               o += align256((size_t)nodes * 128 * 4); // h2
    __bf16* hBbf = xbf;                            // x bf16 copy is dead after layer-1 GEMM
    (void)ws_size; (void)n_in; (void)out_size;

    // degrees (dst shared by all layers) -> reciprocal
    hipMemsetAsync(inv, 0, (size_t)nodes * 4, stream);
    deg_kernel<<<cdiv(E, TPB), TPB, 0, stream>>>(dst, inv, E);
    inv_kernel<<<cdiv(nodes, TPB), TPB, 0, stream>>>(inv, nodes);

    // bf16 transposed weights + bf16 copy of x
    wt_convert<<<cdiv(8192, TPB), TPB, 0, stream>>>(W1l, wt1l, 128, 64);
    wt_convert<<<cdiv(8192, TPB), TPB, 0, stream>>>(W1r, wt1r, 128, 64);
    wt_convert<<<cdiv(8192, TPB), TPB, 0, stream>>>(W2l, wt2l, 64, 128);
    wt_convert<<<cdiv(8192, TPB), TPB, 0, stream>>>(W2r, wt2r, 64, 128);
    wt_convert<<<cdiv(8192, TPB), TPB, 0, stream>>>(W3l, wt3l, 128, 64);
    wt_convert<<<cdiv(8192, TPB), TPB, 0, stream>>>(W3r, wt3r, 128, 64);
    cvt_bf16_kernel<<<cdiv((long long)nodes * 32, TPB), TPB, 0, stream>>>(x, xbf, (long long)nodes * 32);

    const int mTiles = cdiv(nodes, 16);           // 6250, exact

    // ---- layer 1: d=128 -> 64
    hipMemsetAsync(agg, 0, (size_t)nodes * 128 * 4, stream);
    scatter128_kernel<<<cdiv((long long)E * 32, TPB), TPB, 0, stream>>>(x, src, dst, agg, E);
    meanbf_kernel<128><<<cdiv((long long)nodes * 32, TPB), TPB, 0, stream>>>(agg, inv, mbf, nodes);
    sage_gemm_kernel<128, 64><<<cdiv((long long)mTiles * 2 * 32, TPB), TPB, 0, stream>>>(
        mbf, xbf, wt1l, wt1r, b1, hA, hAbf, nodes);

    // ---- layer 2: d=64 -> 128
    hipMemsetAsync(agg, 0, (size_t)nodes * 64 * 4, stream);
    scatter64_kernel<<<cdiv((long long)E * 16, TPB), TPB, 0, stream>>>(hA, src, dst, agg, E);
    meanbf_kernel<64><<<cdiv((long long)nodes * 16, TPB), TPB, 0, stream>>>(agg, inv, mbf, nodes);
    sage_gemm_kernel<64, 128><<<cdiv((long long)mTiles * 4 * 32, TPB), TPB, 0, stream>>>(
        mbf, hAbf, wt2l, wt2r, b2, hB, hBbf, nodes);

    // ---- layer 3: d=128 -> 64  (h3 reuses hA/hAbf; dead after layer-2 GEMM)
    hipMemsetAsync(agg, 0, (size_t)nodes * 128 * 4, stream);
    scatter128_kernel<<<cdiv((long long)E * 32, TPB), TPB, 0, stream>>>(hB, src, dst, agg, E);
    meanbf_kernel<128><<<cdiv((long long)nodes * 32, TPB), TPB, 0, stream>>>(agg, inv, mbf, nodes);
    sage_gemm_kernel<128, 64><<<cdiv((long long)mTiles * 2 * 32, TPB), TPB, 0, stream>>>(
        mbf, hBbf, wt3l, wt3r, b3, hA, hAbf, nodes);

    // ---- head: 64 -> 10
    head_kernel<<<cdiv((long long)nodes * 10, TPB), TPB, 0, stream>>>(hA, Wlin, blin, out, nodes);
}
